// ResidualVectorQuantizer_20650202759519
// MI455X (gfx1250) — matrix operand order repaired and checked by reading, not compiled, not run on hardware
//
#include <hip/hip_runtime.h>
#include <hip/hip_fp16.h>

typedef _Float16 v16h __attribute__((ext_vector_type(16)));
typedef _Float16 h8   __attribute__((ext_vector_type(8)));
typedef _Float16 h4   __attribute__((ext_vector_type(4)));
typedef float    v8f  __attribute__((ext_vector_type(8)));

#define BB 16
#define SS 4096
#define DD 128
#define QQ 4
#define KK 512
#define ROWS (BB * SS)          // 65536
#define PITCH 136               // halves per codebook row in LDS (bank-conflict pad)
#define LDS_CB_BYTES (KK * PITCH * 2)            // 139264
#define LDS_TOTAL_BYTES (LDS_CB_BYTES + KK * 4 + 8 * 16 * 4)  // +c2 +idx = 141824

__global__ __launch_bounds__(256, 1)
void rvq_wmma_kernel(const float* __restrict__ x, const float* __restrict__ cb,
                     float* __restrict__ outq, float* __restrict__ outi,
                     float* __restrict__ outl)
{
    extern __shared__ char smem[];
    _Float16* s_cb  = (_Float16*)smem;                       // [KK][PITCH] f16
    float*    s_c2  = (float*)(smem + LDS_CB_BYTES);         // [KK] exact f32 |c|^2
    int*      s_idx = (int*)(smem + LDS_CB_BYTES + KK * 4);  // [8 waves][16 rows]

    const int tid  = threadIdx.x;
    const int lane = tid & 31;
    const int w    = tid >> 5;
    const int half = lane >> 4;   // 0: lanes 0-15, 1: lanes 16-31
    const int lm   = lane & 15;

    const int rowBase = blockIdx.x * 128 + w * 16;
    const int myRow   = rowBase + lm;

    // Residual registers in WMMA f16 A-matrix layout:
    // r[c][j]: j<8  -> d = c*32 + half*8 + j
    //          j>=8 -> d = c*32 + 16 + half*8 + (j-8)
    float r[4][16];
    {
        const float* xr = x + (size_t)myRow * DD;
        #pragma unroll
        for (int c = 0; c < 4; ++c) {
            const int d0 = c * 32 + half * 8;
            const float4 a = *(const float4*)(xr + d0);
            const float4 b = *(const float4*)(xr + d0 + 4);
            const float4 e = *(const float4*)(xr + d0 + 16);
            const float4 f = *(const float4*)(xr + d0 + 20);
            r[c][0]=a.x;  r[c][1]=a.y;  r[c][2]=a.z;  r[c][3]=a.w;
            r[c][4]=b.x;  r[c][5]=b.y;  r[c][6]=b.z;  r[c][7]=b.w;
            r[c][8]=e.x;  r[c][9]=e.y;  r[c][10]=e.z; r[c][11]=e.w;
            r[c][12]=f.x; r[c][13]=f.y; r[c][14]=f.z; r[c][15]=f.w;
        }
    }

    float lossAcc = 0.0f;

    for (int q = 0; q < QQ; ++q) {
        __syncthreads();  // previous-stage LDS reads done before overwrite
        const float* cbq = cb + (size_t)q * KK * DD;

        // Stage codebook -> LDS (f16) + exact f32 squared norms
        for (int k = tid; k < KK; k += 256) {
            const float* src = cbq + (size_t)k * DD;
            _Float16*    dst = s_cb + k * PITCH;
            float ss = 0.0f;
            #pragma unroll
            for (int d = 0; d < DD; d += 4) {
                const float4 v = *(const float4*)(src + d);
                ss += v.x * v.x + v.y * v.y + v.z * v.z + v.w * v.w;
                h4 hv;
                hv[0] = (_Float16)v.x; hv[1] = (_Float16)v.y;
                hv[2] = (_Float16)v.z; hv[3] = (_Float16)v.w;
                *(h4*)(dst + d) = hv;
            }
            s_c2[k] = ss;
        }
        if (q + 1 < QQ)  // warm next stage's codebook (global_prefetch_b8)
            __builtin_prefetch(cb + (size_t)(q + 1) * KK * DD + (size_t)tid * 256, 0, 0);
        __syncthreads();

        // A operands: f32 residual -> f16, already in A layout; reused by all tiles
        v16h A[4];
        #pragma unroll
        for (int c = 0; c < 4; ++c) {
            #pragma unroll
            for (int j = 0; j < 16; ++j) A[c][j] = (_Float16)r[c][j];
        }

        float minv[8];
        int   mini[8];
        #pragma unroll
        for (int rr = 0; rr < 8; ++rr) { minv[rr] = 3.4e38f; mini[rr] = 0; }

        // Sweep code tiles two at a time: two independent WMMA accumulator
        // chains (same A, different B) so the scheduler can interleave them
        // and fill WMMA latency + WMMA->VALU hazard slots.
        for (int n0 = 0; n0 < KK; n0 += 32) {
            v8f acc0 = {};
            v8f acc1 = {};
            #pragma unroll
            for (int c = 0; c < 4; ++c) {
                // B layout: lane holds code column, K = half*8 + j (j<8), 16 + half*8 + (j-8) (j>=8)
                const _Float16* bp0 = s_cb + (n0 + lm) * PITCH + c * 32 + half * 8;
                const _Float16* bp1 = bp0 + 16 * PITCH;
                v16h B0, B1;
                ((h8*)&B0)[0] = *(const h8*)(bp0);
                ((h8*)&B0)[1] = *(const h8*)(bp0 + 16);
                ((h8*)&B1)[0] = *(const h8*)(bp1);
                ((h8*)&B1)[1] = *(const h8*)(bp1 + 16);
                acc0 = __builtin_amdgcn_wmma_f32_16x16x32_f16(
                    false, A[c], false, B0, (short)0, acc0, false, false);
                acc1 = __builtin_amdgcn_wmma_f32_16x16x32_f16(
                    false, A[c], false, B1, (short)0, acc1, false, false);
            }
            // This lane: codes n0+lm and n0+16+lm; rows M = half*8 + rr.
            // argmin(c2 - 2*cross); r^2 is row-constant, irrelevant to argmin.
            const float c2a = s_c2[n0 + lm];
            const float c2b = s_c2[n0 + 16 + lm];
            const int   code0 = n0 + lm;
            const int   code1 = n0 + 16 + lm;
            #pragma unroll
            for (int rr = 0; rr < 8; ++rr) {
                const float s0 = __builtin_fmaf(-2.0f, acc0[rr], c2a);
                if (s0 < minv[rr]) { minv[rr] = s0; mini[rr] = code0; }
                const float s1 = __builtin_fmaf(-2.0f, acc1[rr], c2b);
                if (s1 < minv[rr]) { minv[rr] = s1; mini[rr] = code1; }
            }
        }

        // Butterfly reduce argmin across the 16-lane group (ties -> lower index)
        #pragma unroll
        for (int m = 1; m < 16; m <<= 1) {
            #pragma unroll
            for (int rr = 0; rr < 8; ++rr) {
                const float ov = __shfl_xor(minv[rr], m, 32);
                const int   oi = __shfl_xor(mini[rr], m, 32);
                if (ov < minv[rr] || (ov == minv[rr] && oi < mini[rr])) {
                    minv[rr] = ov; mini[rr] = oi;
                }
            }
        }

        // Publish per-row indices (lane 0 -> rows 0..7, lane 16 -> rows 8..15)
        if (lm == 0) {
            #pragma unroll
            for (int rr = 0; rr < 8; ++rr) {
                const int row = rowBase + half * 8 + rr;
                s_idx[w * 16 + half * 8 + rr] = mini[rr];
                outi[(size_t)row * QQ + q] = (float)mini[rr];
            }
        }
        __syncthreads();
        const int myCode = s_idx[w * 16 + lm];

        // Exact f32 gather + residual update + commitment loss
        const float* qr = cbq + (size_t)myCode * DD;
        #pragma unroll
        for (int c = 0; c < 4; ++c) {
            const int d0 = c * 32 + half * 8;
            const float4 a = *(const float4*)(qr + d0);
            const float4 b = *(const float4*)(qr + d0 + 4);
            const float4 e = *(const float4*)(qr + d0 + 16);
            const float4 f = *(const float4*)(qr + d0 + 20);
            const float qv[16] = {a.x,a.y,a.z,a.w, b.x,b.y,b.z,b.w,
                                  e.x,e.y,e.z,e.w, f.x,f.y,f.z,f.w};
            #pragma unroll
            for (int j = 0; j < 16; ++j) {
                const float diff = qv[j] - r[c][j];
                lossAcc += diff * diff;
                r[c][j] -= qv[j];
            }
        }
    }

    // quantized = x - residual_final (STE makes output numerically sum of q_steps)
    {
        const float* xr   = x    + (size_t)myRow * DD;
        float*       orow = outq + (size_t)myRow * DD;
        #pragma unroll
        for (int c = 0; c < 4; ++c) {
            const int d0 = c * 32 + half * 8;
            const float4 a = *(const float4*)(xr + d0);
            const float4 b = *(const float4*)(xr + d0 + 4);
            const float4 e = *(const float4*)(xr + d0 + 16);
            const float4 f = *(const float4*)(xr + d0 + 20);
            float4 oa = {a.x - r[c][0],  a.y - r[c][1],  a.z - r[c][2],  a.w - r[c][3]};
            float4 ob = {b.x - r[c][4],  b.y - r[c][5],  b.z - r[c][6],  b.w - r[c][7]};
            float4 oe = {e.x - r[c][8],  e.y - r[c][9],  e.z - r[c][10], e.w - r[c][11]};
            float4 of = {f.x - r[c][12], f.y - r[c][13], f.z - r[c][14], f.w - r[c][15]};
            *(float4*)(orow + d0)      = oa;
            *(float4*)(orow + d0 + 4)  = ob;
            *(float4*)(orow + d0 + 16) = oe;
            *(float4*)(orow + d0 + 20) = of;
        }
    }

    // Loss: wave reduce, one atomic per wave. scale = COMMIT_W / (B*S*D)
    #pragma unroll
    for (int m = 1; m < 32; m <<= 1) lossAcc += __shfl_xor(lossAcc, m, 32);
    if (lane == 0)
        atomicAdd(outl, lossAcc * (0.25f / (float)((size_t)ROWS * DD)));
}

extern "C" void kernel_launch(void* const* d_in, const int* in_sizes, int n_in,
                              void* d_out, int out_size, void* d_ws, size_t ws_size,
                              hipStream_t stream) {
    const float* x  = (const float*)d_in[0];   // [16,4096,128] f32
    const float* cb = (const float*)d_in[1];   // [4,512,128]  f32
    float* outq = (float*)d_out;                         // [B,S,D]
    float* outi = outq + (size_t)ROWS * DD;              // [B,S,Q]
    float* outl = outi + (size_t)ROWS * QQ;              // scalar loss
    hipMemsetAsync(outl, 0, sizeof(float), stream);      // capture-legal
    rvq_wmma_kernel<<<dim3(ROWS / 128), dim3(256), LDS_TOTAL_BYTES, stream>>>(
        x, cb, outq, outi, outl);
}
